// GCN_29841432773036
// MI455X (gfx1250) — matrix operand order repaired and checked by reading, compiled, not run on hardware
//
#include <hip/hip_runtime.h>
#include <hip/hip_bf16.h>
#include <math.h>

#define N_NODES 50000
#define N_EDGES 1600000

typedef __attribute__((ext_vector_type(16))) __bf16    v16bf;
typedef __attribute__((ext_vector_type(8)))  float     v8f;
typedef __attribute__((ext_vector_type(4)))  unsigned  v4u;
typedef __attribute__((ext_vector_type(8)))  int       v8i;
typedef __attribute__((ext_vector_type(4)))  int       v4i;

union BFrag {
  unsigned u[8];
  uint4    q[2];
  v16bf    v;
};

// round-to-nearest-even fp32 -> bf16, packed pair into one dword
__device__ __forceinline__ unsigned pack2_bf16(float a, float b) {
  unsigned ua = __float_as_uint(a);
  unsigned ub = __float_as_uint(b);
  ua += 0x7FFFu + ((ua >> 16) & 1u);
  ub += 0x7FFFu + ((ub >> 16) & 1u);
  return (ua >> 16) | (ub & 0xFFFF0000u);
}

// ---------------- degree / norm kernels ----------------

__global__ void __launch_bounds__(256) k_fill(float* __restrict__ p, int n, float v) {
  int i = blockIdx.x * blockDim.x + threadIdx.x;
  if (i < n) p[i] = v;
}

__global__ void __launch_bounds__(256)
k_deg_edges(const int* __restrict__ dst, float* __restrict__ deg, int ne) {
  int e = blockIdx.x * blockDim.x + threadIdx.x;
  if (e < ne) atomicAdd(&deg[dst[e]], 1.0f);
}

__global__ void __launch_bounds__(256) k_rsqrt_inplace(float* __restrict__ p, int n) {
  int i = blockIdx.x * blockDim.x + threadIdx.x;
  if (i < n) p[i] = rsqrtf(p[i]);
}

// ---------------- weight pre-pack (once, to global scratch) ----------------
// Packs W[KIN,NOUT] (fp32, zero-padded to NPAD cols) into a contiguous-k 32x16 bf16
// B-fragment lane image: fragment (kc,nt): lane<16 holds column n=nt*16+lane,
// K=kc*32+0..15 (2 per dword); lane>=16 holds same column, K=kc*32+16..31.
template<int KIN, int NOUT, int NPAD>
__global__ void __launch_bounds__(256)
k_packW(const float* __restrict__ W, unsigned* __restrict__ Wpk) {
  constexpr int NT = NPAD / 16;
  constexpr int TOT = (KIN / 32) * NT * 32 * 8;
  int idx = blockIdx.x * blockDim.x + threadIdx.x;
  if (idx >= TOT) return;
  int d    = idx & 7;
  int lane = (idx >> 3) & 31;
  int nt   = (idx >> 8) % NT;
  int kc   = (idx >> 8) / NT;
  int k = kc * 32 + ((lane >= 16) ? 16 : 0) + 2 * d;
  int n = nt * 16 + (lane & 15);
  float w0 = (n < NOUT) ? W[(size_t)k * NOUT + n] : 0.0f;
  float w1 = (n < NOUT) ? W[(size_t)(k + 1) * NOUT + n] : 0.0f;
  Wpk[idx] = pack2_bf16(w0, w1);
}

// ---- TDM: copy TOTD dwords from packed global image into LDS (wave 0 only) ----
__device__ __forceinline__ void tdm_copy_to_lds(const unsigned* gsrc, unsigned* lds,
                                                unsigned totd) {
  unsigned long long ga = (unsigned long long)(uintptr_t)gsrc;
  unsigned ldsoff = (unsigned)(uintptr_t)lds;   // LDS aperture: addr[31:0] = LDS byte addr
  v4u g0;
  g0.x = 1u;                                            // count=1, user mode, no gather
  g0.y = ldsoff;                                        // lds_addr
  g0.z = (unsigned)(ga & 0xFFFFFFFFu);                  // global_addr[31:0]
  g0.w = (unsigned)((ga >> 32) & 0x01FFFFFFu) | (2u << 30);  // addr[56:32] | type=2
  v8i g1;
  g1[0] = 0x00020000;                                   // data_size=4B, no flags
  g1[1] = (int)((totd & 0xFFFFu) << 16);                // tensor_dim0[15:0]
  g1[2] = (int)(((totd >> 16) & 0xFFFFu) | (1u << 16)); // tensor_dim0[31:16] | tensor_dim1=1
  g1[3] = (int)((totd & 0xFFFFu) << 16);                // tile_dim0 = totd (<=65535)
  g1[4] = 1;                                            // tile_dim1=1
  g1[5] = (int)totd;                                    // tensor_dim0_stride
  g1[6] = 0;
  g1[7] = 0;
  v4i z4 = {0, 0, 0, 0};
  v8i z8 = {0, 0, 0, 0, 0, 0, 0, 0};
  __builtin_amdgcn_tensor_load_to_lds(g0, g1, z4, z4, z8, 0);   // 6-arg (clang-23) form
  __builtin_amdgcn_s_wait_tensorcnt(0);
}

// ---------------- WMMA GEMM: H[M,NOUT] = f(A[M,KIN]) @ W[KIN,NOUT] ----------------
// f = identity (layer1) or relu(a + bias) (layer2, FUSE=true).
// Packed W image TDM-copied into LDS. Wave computes a 32-row strip (2 A tiles)
// across all N-tiles; block = 8 waves = 256 rows.
// __launch_bounds__(256,1): keep all C tiles in VGPRs (no spills).

template<int KIN, int NOUT, int NPAD, bool FUSE>
__global__ void __launch_bounds__(256, 1)
k_gemm_wmma(const float* __restrict__ A,
            const unsigned* __restrict__ Wpk,
            const float* __restrict__ bias,
            float* __restrict__ Hout) {
  constexpr int KT = KIN / 32;    // k-chunks of 32 (bf16 WMMA depth)
  constexpr int NT = NPAD / 16;   // 16-wide n tiles
  constexpr unsigned TOTD = KT * NT * 32 * 8;
  extern __shared__ unsigned ldsB[];   // TOTD dwords

  const int tid = threadIdx.x;
  if (tid < 32) tdm_copy_to_lds(Wpk, ldsB, TOTD);   // async DMA, wave-0 waits tensorcnt
  __syncthreads();

  const int lane = tid & 31;
  const int wave = tid >> 5;
  const int rsel = lane & 15;   // row within A-tile / column within C-tile
  const int hsel = lane >> 4;   // which K-half (A) / which M-half (C)

  const int m0 = blockIdx.x * 256 + wave * 32;
  if (m0 >= N_NODES) return;    // wave-uniform; no barriers after this point

  int row0 = m0 + rsel;       if (row0 >= N_NODES) row0 = N_NODES - 1;  // clamp:
  int row1 = m0 + 16 + rsel;  if (row1 >= N_NODES) row1 = N_NODES - 1;  // EXEC all-1s
  const float* ar0 = A + (size_t)row0 * KIN;
  const float* ar1 = A + (size_t)row1 * KIN;

  v8f c[2][NT];
#pragma unroll
  for (int rt = 0; rt < 2; ++rt)
#pragma unroll
    for (int t = 0; t < NT; ++t)
      c[rt][t] = (v8f){0.f, 0.f, 0.f, 0.f, 0.f, 0.f, 0.f, 0.f};

  for (int kc = 0; kc < KT; ++kc) {
    if (kc + 1 < KT) {
      __builtin_prefetch(ar0 + (kc + 1) * 32, 0, 1);   // global_prefetch_b8
      __builtin_prefetch(ar1 + (kc + 1) * 32, 0, 1);
    }

    // ---- two A fragments (rows m0.., m0+16..): 16x32 bf16, contiguous k (k = 16*hsel + s) ----
    BFrag af[2];
    const int kb = kc * 32 + hsel * 16;
#pragma unroll
    for (int rt = 0; rt < 2; ++rt) {
      const float* ar = rt ? ar1 : ar0;
#pragma unroll
      for (int j = 0; j < 2; ++j) {         // dwords j*4..j*4+3 <-> K kb+j*8 ..+7
        const float4* ap = (const float4*)(ar + kb + j * 8);
        float4 p0 = ap[0];
        float4 p1 = ap[1];
        float f[8] = {p0.x, p0.y, p0.z, p0.w, p1.x, p1.y, p1.z, p1.w};
        if constexpr (FUSE) {
          const int k = kb + j * 8;
#pragma unroll
          for (int q = 0; q < 8; ++q) {
            float v = f[q] + bias[k + q];
            f[q] = v > 0.f ? v : 0.f;
          }
        }
        af[rt].u[j * 4 + 0] = pack2_bf16(f[0], f[1]);
        af[rt].u[j * 4 + 1] = pack2_bf16(f[2], f[3]);
        af[rt].u[j * 4 + 2] = pack2_bf16(f[4], f[5]);
        af[rt].u[j * 4 + 3] = pack2_bf16(f[6], f[7]);
      }
    }

    // ---- N tiles, ping-pong B fragments so DS loads overlap WMMA ----
    BFrag bf[2];
    {
      const uint4* bp = (const uint4*)&ldsB[((kc * NT + 0) * 32 + lane) * 8];
      bf[0].q[0] = bp[0];
      bf[0].q[1] = bp[1];
    }
#pragma unroll
    for (int nt = 0; nt < NT; ++nt) {
      if (nt + 1 < NT) {
        const uint4* bp = (const uint4*)&ldsB[((kc * NT + nt + 1) * 32 + lane) * 8];
        bf[(nt + 1) & 1].q[0] = bp[0];
        bf[(nt + 1) & 1].q[1] = bp[1];
      }
      c[0][nt] = __builtin_amdgcn_wmma_f32_16x16x32_bf16(
          false, af[0].v, false, bf[nt & 1].v, (short)0, c[0][nt], false, false);
      c[1][nt] = __builtin_amdgcn_wmma_f32_16x16x32_bf16(
          false, af[1].v, false, bf[nt & 1].v, (short)0, c[1][nt], false, false);
    }
  }

  // ---- store C: VGPR r, lane L -> row base + r + (L>=16)*8, col nt*16 + (L&15)
#pragma unroll
  for (int rt = 0; rt < 2; ++rt) {
    const int mb = m0 + rt * 16;
#pragma unroll
    for (int nt = 0; nt < NT; ++nt) {
      const int col = nt * 16 + rsel;
#pragma unroll
      for (int r = 0; r < 8; ++r) {
        const int orow = mb + r + hsel * 8;
        if (orow < N_NODES && col < NOUT)
          Hout[(size_t)orow * NOUT + col] = c[rt][nt][r];
      }
    }
  }
}

// ---------------- aggregation kernels ----------------

// out[i,:] = h[i,:] * dinv[i]^2   (self-loop contribution, initializes out)
template<int C>
__global__ void __launch_bounds__(256)
k_self_init(const float* __restrict__ h,
            const float* __restrict__ dinv,
            float* __restrict__ out) {
  constexpr int C4 = C / 4;
  int i = blockIdx.x * blockDim.x + threadIdx.x;
  int node = i / C4;
  int c4 = i % C4;
  if (node < N_NODES) {
    float w = dinv[node];
    w = w * w;
    float4 v = ((const float4*)(h + (size_t)node * C))[c4];
    float4 o = {v.x * w, v.y * w, v.z * w, v.w * w};
    ((float4*)(out + (size_t)node * C))[c4] = o;
  }
}

// out[dst,:] += h[src,:] * dinv[src]*dinv[dst]
// thread per (edge, float4 chunk): all 32 lanes active for any C.
template<int C>
__global__ void __launch_bounds__(256)
k_scatter_edges(const int* __restrict__ src,
                const int* __restrict__ dst,
                const float* __restrict__ dinv,
                const float* __restrict__ h,
                float* __restrict__ out) {
  constexpr int C4 = C / 4;
  const long long total = (long long)N_EDGES * C4;
  long long t0 = (long long)blockIdx.x * blockDim.x + threadIdx.x;
  const long long stride = (long long)gridDim.x * blockDim.x;
  for (long long t = t0; t < total; t += stride) {
    int e  = (int)(t / C4);
    int c4 = (int)(t % C4);
    int s = src[e];
    int d = dst[e];
    float w = dinv[s] * dinv[d];
    float4 v = ((const float4*)(h + (size_t)s * C))[c4];
    float* op = out + (size_t)d * C + c4 * 4;
    atomicAdd(op + 0, v.x * w);
    atomicAdd(op + 1, v.y * w);
    atomicAdd(op + 2, v.z * w);
    atomicAdd(op + 3, v.w * w);
  }
}

// in-place: a[row,:40] = log_softmax(a[row,:40] + b2), one wave per row
__global__ void __launch_bounds__(256)
k_logsoftmax40(float* __restrict__ a, const float* __restrict__ b2) {
  int gtid = blockIdx.x * blockDim.x + threadIdx.x;
  int lane = gtid & 31;
  int wid = gtid >> 5;
  int nw = (gridDim.x * blockDim.x) >> 5;
  for (int row = wid; row < N_NODES; row += nw) {
    float* r = a + (size_t)row * 40;
    float v1 = r[lane] + b2[lane];
    float v2 = (lane < 8) ? (r[32 + lane] + b2[32 + lane]) : -INFINITY;
    float m = fmaxf(v1, v2);
#pragma unroll
    for (int off = 16; off; off >>= 1) m = fmaxf(m, __shfl_xor(m, off, 32));
    float s = __expf(v1 - m) + ((lane < 8) ? __expf(v2 - m) : 0.f);
#pragma unroll
    for (int off = 16; off; off >>= 1) s += __shfl_xor(s, off, 32);
    float ls = m + __logf(s);
    r[lane] = v1 - ls;
    if (lane < 8) r[32 + lane] = v2 - ls;
  }
}

// ---------------- host launcher ----------------

extern "C" void kernel_launch(void* const* d_in, const int* in_sizes, int n_in,
                              void* d_out, int out_size, void* d_ws, size_t ws_size,
                              hipStream_t stream) {
  const float* x  = (const float*)d_in[0];
  const int*   ei = (const int*)d_in[1];
  const float* W1 = (const float*)d_in[2];
  const float* b1 = (const float*)d_in[3];
  const float* W2 = (const float*)d_in[4];
  const float* b2 = (const float*)d_in[5];
  const int* srcE = ei;             // edge_index[0]
  const int* dstE = ei + N_EDGES;   // edge_index[1]

  constexpr unsigned TOTD1 = (512 / 32) * (128 / 16) * 32 * 8;   // 32768 dwords, 128 KB
  constexpr unsigned TOTD2 = (128 / 32) * (48 / 16) * 32 * 8;    //  3072 dwords,  12 KB

  float* ws   = (float*)d_ws;
  float* dinv = ws;                                    // 50000 (padded to 65536)
  float* h1   = ws + 65536;                            // 50000 x 128
  float* a1   = h1 + (size_t)N_NODES * 128;            // 50000 x 128
  float* h2   = a1 + (size_t)N_NODES * 128;            // 50000 x 40
  unsigned* wpk1 = (unsigned*)(h2 + (size_t)N_NODES * 40);  // packed W1 image
  unsigned* wpk2 = wpk1 + TOTD1;                            // packed W2 image
  float* out  = (float*)d_out;                         // 50000 x 40

  const int T = 256;

  // deg (incl. self-loop) -> dinv = deg^-1/2 ; pack weights once
  k_fill<<<(N_NODES + T - 1) / T, T, 0, stream>>>(dinv, N_NODES, 1.0f);
  k_deg_edges<<<(N_EDGES + T - 1) / T, T, 0, stream>>>(dstE, dinv, N_EDGES);
  k_rsqrt_inplace<<<(N_NODES + T - 1) / T, T, 0, stream>>>(dinv, N_NODES);
  k_packW<512, 128, 128><<<(TOTD1 + T - 1) / T, T, 0, stream>>>(W1, wpk1);
  k_packW<128, 40, 48><<<(TOTD2 + T - 1) / T, T, 0, stream>>>(W2, wpk2);

  // layer 1: h1 = x @ W1  (bf16 WMMA, fp32 acc; W via TDM -> LDS)
  {
    int grid = (N_NODES + 255) / 256;
    k_gemm_wmma<512, 128, 128, false>
        <<<grid, T, TOTD1 * 4, stream>>>(x, wpk1, nullptr, h1);
  }
  // aggregate: a1 = D^-1/2 A D^-1/2 h1
  k_self_init<128><<<(N_NODES * 32 + T - 1) / T, T, 0, stream>>>(h1, dinv, a1);
  k_scatter_edges<128><<<4096, T, 0, stream>>>(srcE, dstE, dinv, h1, a1);

  // layer 2: h2 = relu(a1 + b1) @ W2   (N padded 40 -> 48)
  {
    int grid = (N_NODES + 255) / 256;
    k_gemm_wmma<128, 40, 48, true>
        <<<grid, T, TOTD2 * 4, stream>>>(a1, wpk2, b1, h2);
  }
  // aggregate into d_out, then + b2 and log_softmax in place
  k_self_init<40><<<(N_NODES * 10 + T - 1) / T, T, 0, stream>>>(h2, dinv, out);
  k_scatter_edges<40><<<4096, T, 0, stream>>>(srcE, dstE, dinv, h2, out);
  k_logsoftmax40<<<1024, T, 0, stream>>>(out, b2);
}
